// stan_2d_model_56770877718606
// MI455X (gfx1250) — matrix operand order repaired and checked by reading, not compiled, not run on hardware
//
#include <hip/hip_runtime.h>
#include <hip/hip_bf16.h>

typedef __bf16 bf16;
typedef __attribute__((ext_vector_type(16))) __bf16 v16bf;
typedef __attribute__((ext_vector_type(8)))  float   v8f;
typedef __attribute__((ext_vector_type(4)))  unsigned int u32x4;
typedef __attribute__((ext_vector_type(8)))  int     i32x8;
typedef __attribute__((ext_vector_type(4)))  int     i32x4;

#define B_    256
#define T_    32
#define F_    128
#define H_    128
#define NF_   64
#define OH_   31
#define OW_   255
#define OHW_  7905      // 31*255
#define FLAT_ 505920    // 64*31*255
#define FW_   256       // feat width = 2*F

__device__ __forceinline__ v8f wmma_bf16(v16bf a, v16bf b, v8f c) {
  // 8 args: (neg_a, A, neg_b, B, c_mod, C, reuse_a, reuse_b)
  return __builtin_amdgcn_wmma_f32_16x16x32_bf16(false, a, false, b, (short)0, c, false, false);
}

// ---------------- Kernel 0: zero the split-K accumulator ----------------
__global__ void k_zero(float* p, int n) {
  int i = blockIdx.x * blockDim.x + threadIdx.x;
  if (i < n) p[i] = 0.f;
}

// ---------------- Kernel 1: attention + context + feat ----------------
// One workgroup (256 thr = 8 waves) per batch element.
__global__ void __launch_bounds__(256) k_attn(const float* __restrict__ X,
    const float* __restrict__ Wg, const float* __restrict__ Ug,
    const float* __restrict__ Vg, float* __restrict__ feat)
{
  __shared__ float sXf[T_ * F_];   // 16 KB
  __shared__ bf16  sXb[T_ * F_];   // 8 KB
  __shared__ float sxw[T_ * H_];   // 16 KB
  __shared__ float sxu[T_ * H_];   // 16 KB
  __shared__ float sS[T_ * T_];    // 4 KB
  __shared__ float sV[H_];
  const int tid = threadIdx.x;
  const int b   = blockIdx.x;
  const float* Xb = X + b * T_ * F_;
  for (int e = tid; e < T_ * F_; e += 256) { float v = Xb[e]; sXf[e] = v; sXb[e] = (bf16)v; }
  if (tid < H_) sV[tid] = Vg[tid];
  __syncthreads();

  const int lane = tid & 31, wave = tid >> 5;
  const int m = lane & 15, half = lane >> 4;

  // xw = X@W, xu = X@U : 32 output tiles of 16x16, 4 per wave, K=128 in 4 WMMA steps
  for (int q = 0; q < 4; q++) {
    int t = wave * 4 + q;
    const float* Bsrc = (t < 16) ? Wg : Ug;
    float* Dst = (t < 16) ? sxw : sxu;
    int rem = t & 15;
    int mt = rem >> 3, nt = rem & 7;
    v8f acc = {};
    for (int kk = 0; kk < 4; kk++) {
      int kb = kk * 32;
      const bf16* ar = sXb + (mt * 16 + m) * F_ + kb;
      v16bf a, bb;
#pragma unroll
      for (int e = 0; e < 8; e++) a[e]     = ar[16 * half + e];
#pragma unroll
      for (int e = 0; e < 8; e++) a[8 + e] = ar[16 * half + 8 + e];
      int n = nt * 16 + m;
#pragma unroll
      for (int e = 0; e < 16; e++) bb[e] = (bf16)Bsrc[(kb + 16 * half + e) * H_ + n];
      acc = wmma_bf16(a, bb, acc);
    }
#pragma unroll
    for (int r = 0; r < 8; r++)
      Dst[(mt * 16 + 8 * half + r) * H_ + nt * 16 + m] = acc[r];
  }
  __syncthreads();

  // scores[i][j] = sum_h tanh(xw[i,h]+xu[j,h]) * V[h]
  for (int p = tid; p < T_ * T_; p += 256) {
    int i = p >> 5, j = p & 31;
    float s = 0.f;
    for (int h = 0; h < H_; h++) s += tanhf(sxw[i * H_ + h] + sxu[j * H_ + h]) * sV[h];
    sS[p] = s;
  }
  __syncthreads();

  // softmax per row, then zero the diagonal (mask applied after softmax)
  if (tid < T_) {
    float mx = -1e30f;
    for (int j = 0; j < T_; j++) mx = fmaxf(mx, sS[tid * T_ + j]);
    float sum = 0.f;
    for (int j = 0; j < T_; j++) { float e = __expf(sS[tid * T_ + j] - mx); sS[tid * T_ + j] = e; sum += e; }
    float inv = 1.f / sum;
    for (int j = 0; j < T_; j++) sS[tid * T_ + j] *= inv;
    sS[tid * T_ + tid] = 0.f;
  }
  __syncthreads();

  // c = alpha_masked @ X ; feat = [X, c]
  float* fb = feat + b * T_ * FW_;
  for (int q = tid; q < T_ * F_; q += 256) {
    int t = q >> 7, f = q & 127;
    float c = 0.f;
    for (int j = 0; j < T_; j++) c += sS[t * T_ + j] * sXf[j * F_ + f];
    fb[t * FW_ + f]      = sXf[t * F_ + f];
    fb[t * FW_ + F_ + f] = c;
  }
}

// ---------------- Kernel 2: fused conv + big GEMM (flat @ w1) ----------------
// grid (255 K-chunks, 2 N-slabs of 128), 512 thr = 16 waves.
// Wave w owns M rows [16w,16w+16) x the full 128-col slab (8 WMMA tiles).
// w1 is streamed from HBM exactly once via double-buffered TDM DMA into LDS.
__global__ void __launch_bounds__(512) k_bigmm(const float* __restrict__ feat,
    const float* __restrict__ cw, const float* __restrict__ cb,
    const float* __restrict__ w1, float* __restrict__ acc)
{
  __shared__ float sW32[2][32 * 128];   // double-buffered fp32 w1 tile (2 x 16 KB)
  __shared__ bf16  sB[2][128 * 40];     // double-buffered transposed bf16 panel (2 x 10 KB)
  __shared__ float scw[NF_ * 4];
  __shared__ float scb[NF_];
  const int tid = threadIdx.x;
  if (tid < NF_ * 4) scw[tid] = cw[tid];
  if (tid < NF_)     scb[tid] = cb[tid];
  const int lane = tid & 31, wave = tid >> 5;
  const int m = lane & 15, half = lane >> 4;
  const int n0 = blockIdx.y * 128;
  const int kstep0 = blockIdx.x * 62;       // 255*62*32 == FLAT
  v8f accv[8] = {{}, {}, {}, {}, {}, {}, {}, {}};
  const int brow = wave * 16 + m;           // batch row feeding this lane's A fragment
  const float* fb = feat + brow * (T_ * FW_);

  // Issue a TDM DMA of the 32x128 fp32 w1 tile for k-step s into buffer buf.
  auto tdm_issue = [&](int s, int buf) {
    unsigned long long k0 = (unsigned long long)(kstep0 + s) * 32ull;
    unsigned long long ga = (unsigned long long)w1 + (k0 * 256ull + (unsigned long long)n0) * 4ull;
    unsigned int ldsaddr = (unsigned int)(unsigned long long)&sW32[buf][0];
    u32x4 g0; i32x8 g1; i32x4 g2 = {0, 0, 0, 0}, g3 = {0, 0, 0, 0};
    i32x8 g4 = {0, 0, 0, 0, 0, 0, 0, 0};
    g0[0] = 1u;                                           // count=1, normal descriptor
    g0[1] = ldsaddr;                                      // lds_addr
    g0[2] = (unsigned int)ga;                             // global_addr[31:0]
    g0[3] = (unsigned int)((ga >> 32) & 0x01FFFFFFull) | 0x80000000u; // addr[56:32] | type=2
    g1[0] = (int)0x00020000u;                             // data_size = 4B
    g1[1] = (int)(256u << 16);                            // tensor_dim0 = 256 (low16)
    g1[2] = (int)((unsigned)(FLAT_ & 0xFFFF) << 16);      // dim0 hi16 (0) | tensor_dim1 low16
    g1[3] = (int)((unsigned)(FLAT_ >> 16) | (128u << 16)); // tensor_dim1 hi16 | tile_dim0=128
    g1[4] = 32;                                           // tile_dim1 = 32 rows
    g1[5] = 256;                                          // tensor_dim0_stride = 256
    g1[6] = 0; g1[7] = 0;
    __builtin_amdgcn_tensor_load_to_lds(g0, g1, g2, g3, g4, 0);
  };

  if (tid < 32) tdm_issue(0, 0);          // prologue: tile 0 in flight

  for (int s = 0; s < 62; s++) {
    const int cur = s & 1;
    long long k0 = (long long)(kstep0 + s) * 32;
    if (tid < 32) __builtin_amdgcn_s_wait_tensorcnt(0);  // tile s landed (wave 0)
    __syncthreads();          // sW32[cur] visible; compute s-1 done; sB[cur] reusable
    if (tid < 32 && s + 1 < 62) tdm_issue(s + 1, cur ^ 1);   // overlap DMA with work
    if (tid == 0 && s + 1 < 62)
      __builtin_prefetch(w1 + ((long long)(kstep0 + s + 1) * 32 * 256 + n0), 0, 0);

    // convert fp32 tile -> transposed bf16 panel [n][k] (k-pairs -> packed writes)
    for (int p = tid; p < 16 * 128; p += 512) {
      int k2 = p & 15, n = p >> 4;
      sB[cur][n * 40 + 2 * k2]     = (bf16)sW32[cur][(2 * k2) * 128 + n];
      sB[cur][n * 40 + 2 * k2 + 1] = (bf16)sW32[cur][(2 * k2 + 1) * 128 + n];
    }
    __syncthreads();          // sB[cur] ready

    // Build A fragment on the fly: fused 2x2 conv + bias + relu from feat (L2-resident)
    v16bf a;
#pragma unroll
    for (int e = 0; e < 16; e++) {
      int kk = 16 * half + e;
      unsigned int kg = (unsigned int)k0 + (unsigned int)kk;
      unsigned int o = kg / OHW_;
      unsigned int r = kg - o * OHW_;
      unsigned int i = r / OW_;
      unsigned int j = r - i * OW_;
      const float* fp = fb + i * FW_ + j;
      float v = scb[o] + scw[o * 4 + 0] * fp[0]       + scw[o * 4 + 1] * fp[1]
                       + scw[o * 4 + 2] * fp[FW_]     + scw[o * 4 + 3] * fp[FW_ + 1];
      a[e] = (bf16)fmaxf(v, 0.f);
    }
#pragma unroll
    for (int nt = 0; nt < 8; nt++) {
      v16bf bb;
      const bf16* cbase = sB[cur] + (nt * 16 + m) * 40 + 16 * half;
#pragma unroll
      for (int e = 0; e < 16; e++) bb[e] = cbase[e];
      accv[nt] = wmma_bf16(a, bb, accv[nt]);
    }
  }

  // split-K accumulate into the 256x256 output
#pragma unroll
  for (int nt = 0; nt < 8; nt++) {
    int col = n0 + nt * 16 + m;
#pragma unroll
    for (int r = 0; r < 8; r++) {
      int row = wave * 16 + 8 * half + r;
      atomicAdd(&acc[row * 256 + col], accv[nt][r]);
    }
  }
}

// ---------------- Kernel 3a: h1 = relu(acc + b1) ----------------
__global__ void k_h1(const float* __restrict__ acc, const float* __restrict__ b1,
                     float* __restrict__ h1) {
  int i = blockIdx.x * blockDim.x + threadIdx.x;
  if (i < 256 * 256) h1[i] = fmaxf(acc[i] + b1[i & 255], 0.f);
}

// ---------------- Kernel 3b: h2 = relu(h1@w2+b2); out = h2@w3+b3 ----------------
__global__ void __launch_bounds__(512) k_head(const float* __restrict__ h1,
    const float* __restrict__ w2, const float* __restrict__ b2,
    const float* __restrict__ w3, const float* __restrict__ b3,
    float* __restrict__ out)
{
  __shared__ bf16  sW2[32 * 256];   // transposed w2 [n][k], 16 KB
  __shared__ float sH2[256 * 32];   // 32 KB
  const int tid = threadIdx.x;
  for (int e = tid; e < 256 * 32; e += 512) {
    int k = e >> 5, n = e & 31;
    sW2[n * 256 + k] = (bf16)w2[e];
  }
  __syncthreads();
  const int lane = tid & 31, wave = tid >> 5;
  const int m = lane & 15, half = lane >> 4;
  v8f acc2[2] = {{}, {}};
  const float* ar0 = h1 + (wave * 16 + m) * 256;
  for (int kk = 0; kk < 8; kk++) {
    v16bf a;
    const float* ar = ar0 + kk * 32;
#pragma unroll
    for (int e = 0; e < 8; e++) a[e]     = (bf16)ar[16 * half + e];
#pragma unroll
    for (int e = 0; e < 8; e++) a[8 + e] = (bf16)ar[16 * half + 8 + e];
#pragma unroll
    for (int nt = 0; nt < 2; nt++) {
      v16bf bb;
      const bf16* cbase = sW2 + (nt * 16 + m) * 256 + kk * 32 + 16 * half;
#pragma unroll
      for (int e = 0; e < 16; e++) bb[e] = cbase[e];
      acc2[nt] = wmma_bf16(a, bb, acc2[nt]);
    }
  }
#pragma unroll
  for (int nt = 0; nt < 2; nt++) {
    int col = nt * 16 + m;
    float bias = b2[col];
#pragma unroll
    for (int r = 0; r < 8; r++) {
      int row = wave * 16 + 8 * half + r;
      sH2[row * 32 + col] = fmaxf(acc2[nt][r] + bias, 0.f);
    }
  }
  __syncthreads();
  if (tid < 256) {
    float o0 = b3[0], o1 = b3[1];
    for (int j = 0; j < 32; j++) {
      float h = sH2[tid * 32 + j];
      o0 += h * w3[j * 2 + 0];
      o1 += h * w3[j * 2 + 1];
    }
    out[tid * 2 + 0] = o0;
    out[tid * 2 + 1] = o1;
  }
}

// ---------------- launch ----------------
extern "C" void kernel_launch(void* const* d_in, const int* in_sizes, int n_in,
                              void* d_out, int out_size, void* d_ws, size_t ws_size,
                              hipStream_t stream) {
  const float* X  = (const float*)d_in[0];
  const float* aW = (const float*)d_in[1];
  const float* aU = (const float*)d_in[2];
  const float* aV = (const float*)d_in[3];
  const float* cw = (const float*)d_in[4];
  const float* cb = (const float*)d_in[5];
  const float* w1 = (const float*)d_in[6];
  const float* b1 = (const float*)d_in[7];
  const float* w2 = (const float*)d_in[8];
  const float* b2 = (const float*)d_in[9];
  const float* w3 = (const float*)d_in[10];
  const float* b3 = (const float*)d_in[11];

  float* feat = (float*)d_ws;                       // 256*32*256 = 2 Mi floats (8 MB)
  float* acc  = feat + (size_t)256 * 32 * 256;      // 256*256 floats
  float* h1   = acc + 256 * 256;                    // 256*256 floats

  k_zero <<<256, 256, 0, stream>>>(acc, 256 * 256);
  k_attn <<<256, 256, 0, stream>>>(X, aW, aU, aV, feat);
  k_bigmm<<<dim3(255, 2), 512, 0, stream>>>(feat, cw, cb, w1, acc);
  k_h1   <<<256, 256, 0, stream>>>(acc, b1, h1);
  k_head <<<1, 512, 0, stream>>>(h1, w2, b2, w3, b3, (float*)d_out);
}
